// ConstrainedLqr_60550448939589
// MI455X (gfx1250) — compile-verified, matrix operand order checked by reading, not run on hardware
//
#include <hip/hip_runtime.h>

// ---------------- problem dimensions ----------------
#define NXd   12
#define NUd   6
#define NVARS 354      // NS*NX + (NS-1)*NU
#define NEQd  240
#define NINd  228
#define NPd   368      // NVARS padded to 16 (23 tiles)
#define EPd   240
#define IPK   256      // NIN padded to 32 for WMMA K loop
#define TPd   608      // NPd + EPd = 19*32
#define NBLK  19
#define HT    23       // NPd/16
#define NBATCH 32
#define SIGMA_F 0.1f
#define EPS_F   1e-8f
#define DELTA_F 1e-6f
#define MAXIT   20
#define BIGF    1e9f

// ---------------- per-batch workspace layout (float offsets) ----------------
#define OFF_KKT 0
#define OFF_RHS 369664
#define OFF_X   370272
#define OFF_S   370640
#define OFF_Z   370880
#define OFF_Y   371120
#define OFF_QD  371360
#define OFF_R3  371728
#define OFF_T   371968
#define OFF_D   372208
#define OFF_IL  372464
#define OFF_IU  373488
#define WS_STRIDE 374512   // floats per batch (~1.43 MB); x32 ~ 48 MB (L2-resident)

typedef __attribute__((ext_vector_type(16))) __bf16          v16bf;
typedef __attribute__((ext_vector_type(16))) unsigned short  v16us;
typedef __attribute__((ext_vector_type(8)))  float           v8f;

struct Frag { v16us hi; v16us lo; };

// split f32 into bf16 hi + bf16 lo
__device__ __forceinline__ void bsplit(float x, unsigned short &h, unsigned short &l) {
  unsigned u = __builtin_bit_cast(unsigned, x);
  h = (unsigned short)(u >> 16);
  float fh = __builtin_bit_cast(float, u & 0xffff0000u);
  float r  = x - fh;
  l = (unsigned short)(__builtin_bit_cast(unsigned, r) >> 16);
}

// CDNA5 16-bit WMMA K pattern: VGPR v holds K = base(v)+{0,1}; lane half adds 8.
__device__ __forceinline__ int kpat(int v, int half) {
  return ((v < 4) ? 0 : 16) + 8 * half + 2 * (v & 3);
}

// ---- unguarded fragment loaders (all addresses in-bounds by construction) ----

// A-frag (16x32) from LDS tile laid out [k][16m]
__device__ __forceinline__ Frag load_a_lds16(const float* sm, float sgn, int lane) {
  Frag f; int half = lane >> 4; int m = lane & 15;
  #pragma unroll
  for (int v = 0; v < 8; ++v) {
    int k0 = kpat(v, half);
    #pragma unroll
    for (int d = 0; d < 2; ++d) {
      float x = sgn * sm[(k0 + d) * 16 + m];
      unsigned short h, l; bsplit(x, h, l);
      f.hi[2*v+d] = h; f.lo[2*v+d] = l;
    }
  }
  return f;
}

// A-frag (16x32) from LDS tile laid out [m][32k]
__device__ __forceinline__ Frag load_a_lds32(const float* sm, float sgn, int lane) {
  Frag f; int half = lane >> 4; int m = lane & 15;
  #pragma unroll
  for (int v = 0; v < 8; ++v) {
    int k0 = kpat(v, half);
    #pragma unroll
    for (int d = 0; d < 2; ++d) {
      float x = sgn * sm[m * 32 + k0 + d];
      unsigned short h, l; bsplit(x, h, l);
      f.hi[2*v+d] = h; f.lo[2*v+d] = l;
    }
  }
  return f;
}

// B-frag (32x16) from LDS panel laid out [k][NPd], columns col0..col0+15
__device__ __forceinline__ Frag load_b_ldsp(const float* sm, int col0, int lane) {
  Frag f; int half = lane >> 4; int n = col0 + (lane & 15);
  #pragma unroll
  for (int v = 0; v < 8; ++v) {
    int k0 = kpat(v, half);
    #pragma unroll
    for (int d = 0; d < 2; ++d) {
      float x = sm[(k0 + d) * NPd + n];
      unsigned short h, l; bsplit(x, h, l);
      f.hi[2*v+d] = h; f.lo[2*v+d] = l;
    }
  }
  return f;
}

// A-frag (16x32) from global row-major: A[m][k] = src[(row0+m)*ld + kbase+k]
__device__ __forceinline__ Frag load_a_g(const float* src, int ld, int row0, int kbase,
                                         float sgn, int lane) {
  Frag f; int half = lane >> 4; int m = row0 + (lane & 15);
  #pragma unroll
  for (int v = 0; v < 8; ++v) {
    int k0 = kpat(v, half);
    #pragma unroll
    for (int d = 0; d < 2; ++d) {
      float x = sgn * src[(size_t)m * ld + kbase + k0 + d];
      unsigned short h, l; bsplit(x, h, l);
      f.hi[2*v+d] = h; f.lo[2*v+d] = l;
    }
  }
  return f;
}

// B-frag (32x16) from global row-major: B[k][n] = src[(kbase+k)*ld + col0+n]
__device__ __forceinline__ Frag load_b_g(const float* src, int ld, int kbase, int col0,
                                         int lane) {
  Frag f; int half = lane >> 4; int n = col0 + (lane & 15);
  #pragma unroll
  for (int v = 0; v < 8; ++v) {
    int k0 = kpat(v, half);
    #pragma unroll
    for (int d = 0; d < 2; ++d) {
      float x = src[(size_t)(kbase + k0 + d) * ld + n];
      unsigned short h, l; bsplit(x, h, l);
      f.hi[2*v+d] = h; f.lo[2*v+d] = l;
    }
  }
  return f;
}

// 3-product bf16-split WMMA: C += Ahi*Bhi + Ahi*Blo + Alo*Bhi  (f32 accumulate)
__device__ __forceinline__ v8f wmma3(const Frag& a, const Frag& b, v8f c) {
  v16bf ah = __builtin_bit_cast(v16bf, a.hi);
  v16bf al = __builtin_bit_cast(v16bf, a.lo);
  v16bf bh = __builtin_bit_cast(v16bf, b.hi);
  v16bf bl = __builtin_bit_cast(v16bf, b.lo);
  c = __builtin_amdgcn_wmma_f32_16x16x32_bf16(false, ah, false, bh, (short)0, c, false, false);
  c = __builtin_amdgcn_wmma_f32_16x16x32_bf16(false, ah, false, bl, (short)0, c, false, false);
  c = __builtin_amdgcn_wmma_f32_16x16x32_bf16(false, al, false, bh, (short)0, c, false, false);
  return c;
}

__device__ __forceinline__ float* wsp(float* ws, int b, int off) {
  return ws + (size_t)b * WS_STRIDE + off;
}

// CDNA5 async global->LDS copy of one dword (ASYNCcnt-tracked)
__device__ __forceinline__ void async_b32(const float* gsrc, float* ldst) {
  unsigned loff = (unsigned)(size_t)(void*)ldst;   // generic LDS ptr: low 32 bits = LDS offset
  asm volatile("global_load_async_to_lds_b32 %0, %1, off"
               :: "v"(loff), "v"(gsrc) : "memory");
}
__device__ __forceinline__ void wait_async0() {
  asm volatile("s_wait_asynccnt 0x0" ::: "memory");
}

// ---------------- K0: init state + qdiag ----------------
__global__ void k_init(const float* __restrict__ w, float* ws) {
  int b = blockIdx.x, tid = threadIdx.x;
  float* X  = wsp(ws, b, OFF_X);
  float* S  = wsp(ws, b, OFF_S);
  float* Z  = wsp(ws, b, OFF_Z);
  float* Y  = wsp(ws, b, OFF_Y);
  float* QD = wsp(ws, b, OFF_QD);
  float* Dv = wsp(ws, b, OFF_D);
  const float* wb = w + b * (NXd/2 + NUd + NXd);   // 24 entries
  for (int i = tid; i < NPd; i += blockDim.x) {
    float q = 0.f;
    if (i < 342) { int j = i % 18; q = (j < 6) ? EPS_F : (wb[j - 6] + EPS_F); }
    else if (i < NVARS) { q = 10000.f + (wb[12 + (i - 342)] + EPS_F); }
    QD[i] = q;
    X[i]  = 0.f;
  }
  for (int k = tid; k < 240; k += blockDim.x) { S[k] = 1.f; Z[k] = 1.f; Y[k] = 0.f; }
  for (int k = tid; k < IPK; k += blockDim.x) Dv[k] = 0.f;
}

// ---------------- K1: residuals, D, t, rhs ----------------
__global__ void k_resid(const float* __restrict__ A, const float* __restrict__ bvec,
                        const float* __restrict__ G, const float* __restrict__ h,
                        float* ws) {
  int b = blockIdx.x, tid = threadIdx.x;
  float* X  = wsp(ws, b, OFF_X);
  float* S  = wsp(ws, b, OFF_S);
  float* Z  = wsp(ws, b, OFF_Z);
  float* Y  = wsp(ws, b, OFF_Y);
  float* QD = wsp(ws, b, OFF_QD);
  float* R3 = wsp(ws, b, OFF_R3);
  float* Tv = wsp(ws, b, OFF_T);
  float* Dv = wsp(ws, b, OFF_D);
  float* RHS = wsp(ws, b, OFF_RHS);
  const float* Ab = A + (size_t)b * NEQd * NVARS;
  const float* Gb = G + (size_t)b * NINd * NVARS;
  __shared__ float red[256];
  __shared__ float muv;
  float p = 0.f;
  for (int k = tid; k < NINd; k += 256) p += S[k] * Z[k];
  red[tid] = p; __syncthreads();
  for (int o = 128; o > 0; o >>= 1) { if (tid < o) red[tid] += red[tid + o]; __syncthreads(); }
  if (tid == 0) muv = red[0] / (float)NINd;
  __syncthreads();
  float mu = muv;
  for (int k = tid; k < NINd; k += 256) {
    const float* gr = Gb + (size_t)k * NVARS;
    float gx = 0.f;
    for (int i = 0; i < NVARS; ++i) gx += gr[i] * X[i];
    R3[k] = gx + S[k] - h[(size_t)b * NINd + k];
  }
  for (int e = tid; e < NEQd; e += 256) {
    const float* ar = Ab + (size_t)e * NVARS;
    float ax = 0.f;
    for (int i = 0; i < NVARS; ++i) ax += ar[i] * X[i];
    RHS[NPd + e] = bvec[(size_t)b * NEQd + e] - ax;   // -r2
  }
  __syncthreads();
  for (int k = tid; k < NINd; k += 256) {
    float s = S[k], z = Z[k];
    float r4 = s * z - SIGMA_F * mu;
    Dv[k] = z / s;
    Tv[k] = (z * R3[k] - r4) / s;
  }
  __syncthreads();
  for (int i = tid; i < NPd; i += 256) {
    if (i < NVARS) {
      float aty = 0.f;
      for (int e = 0; e < NEQd; ++e) aty += Ab[(size_t)e * NVARS + i] * Y[e];
      float gtz = 0.f, gtt = 0.f;
      for (int k = 0; k < NINd; ++k) {
        float g = Gb[(size_t)k * NVARS + i];
        gtz += g * Z[k]; gtt += g * Tv[k];
      }
      RHS[i] = -(QD[i] * X[i] + aty + gtz) - gtt;
    } else RHS[i] = 0.f;
  }
}

// ---------------- K3: H = (D*G)^T G + diag(qd+delta), LDS-staged WMMA ----------------
// One block per 16-row strip of H; 8 waves sweep all 23 column tiles.
// Accumulators fully unrolled (compile-time q) so they stay in VGPRs.
__global__ void k_buildH(const float* __restrict__ G, float* ws) {
  int b = blockIdx.y;
  int i0 = blockIdx.x * 16;
  int tid = threadIdx.x, wave = tid >> 5, lane = tid & 31;
  const float* Gb = G + (size_t)b * NINd * NVARS;
  float* KK = wsp(ws, b, OFF_KKT);
  const float* Dv = wsp(ws, b, OFF_D);
  const float* QD = wsp(ws, b, OFF_QD);
  __shared__ float Apan[32 * 16];      // (D*G)^T tile: [k][m]
  __shared__ float Bpan[32 * NPd];     // raw G K-panel: [k][col], zero-padded
  for (int t = tid; t < 32 * NPd; t += 256) Bpan[t] = 0.f;
  __syncthreads();

  v8f acc0, acc1, acc2;
  #pragma unroll
  for (int r = 0; r < 8; ++r) { acc0[r] = 0.f; acc1[r] = 0.f; acc2[r] = 0.f; }
  int jt0 = wave, jt1 = wave + 8, jt2 = wave + 16;

  for (int kb = 0; kb < IPK; kb += 32) {
    int nval = (NINd - kb < 32) ? (NINd - kb) : 32;
    // stage A tile (guards folded into staging; D scale applied on A side)
    for (int t = tid; t < 32 * 16; t += 256) {
      int k = t >> 4, m = t & 15;
      int kk = kb + k;
      float v = 0.f;
      if (kk < NINd && (i0 + m) < NVARS) v = Dv[kk] * Gb[(size_t)kk * NVARS + i0 + m];
      Apan[t] = v;
    }
    // async-stage raw G panel rows (interior only; pad already zero)
    int total = nval * NVARS;
    for (int idx = tid; idx < total; idx += 256) {
      int k = idx / NVARS, c = idx - k * NVARS;
      async_b32(Gb + (size_t)(kb + k) * NVARS + c, &Bpan[k * NPd + c]);
    }
    wait_async0();
    __syncthreads();

    Frag a = load_a_lds16(Apan, 1.f, lane);
    {
      Frag bb = load_b_ldsp(Bpan, jt0 * 16, lane);
      acc0 = wmma3(a, bb, acc0);
    }
    {
      Frag bb = load_b_ldsp(Bpan, jt1 * 16, lane);
      acc1 = wmma3(a, bb, acc1);
    }
    if (jt2 < HT) {                      // wave-uniform
      Frag bb = load_b_ldsp(Bpan, jt2 * 16, lane);
      acc2 = wmma3(a, bb, acc2);
    }
    __syncthreads();   // protect LDS before next staging
  }

  int half = lane >> 4;
  #pragma unroll
  for (int q = 0; q < 3; ++q) {
    int jt = (q == 0) ? jt0 : (q == 1) ? jt1 : jt2;
    if (jt >= HT) continue;
    int col = jt * 16 + (lane & 15);
    #pragma unroll
    for (int r = 0; r < 8; ++r) {
      int row = i0 + r + 8 * half;
      float v = (q == 0) ? acc0[r] : (q == 1) ? acc1[r] : acc2[r];
      if (row == col) v += (row < NVARS) ? (QD[row] + DELTA_F) : 1.f;  // pad rows -> identity
      KK[(size_t)row * TPd + col] = v;
    }
  }
}

// ---------------- K4: fill A / A^T borders and regularized zero block ----------------
__global__ void k_fill(const float* __restrict__ A, float* ws) {
  int b = blockIdx.y;
  int idx = blockIdx.x * 256 + threadIdx.x;
  const int R1 = NPd * EPd;
  const int TOT = R1 + EPd * TPd;
  if (idx >= TOT) return;
  const float* Ab = A + (size_t)b * NEQd * NVARS;
  float* KK = wsp(ws, b, OFF_KKT);
  if (idx < R1) {
    int r = idx / EPd, e = idx % EPd;
    KK[(size_t)r * TPd + NPd + e] = (r < NVARS) ? Ab[(size_t)e * NVARS + r] : 0.f;
  } else {
    int j = idx - R1;
    int r = NPd + j / TPd, c = j % TPd;
    float v;
    if (c < NPd) v = (c < NVARS) ? Ab[(size_t)(r - NPd) * NVARS + c] : 0.f;
    else         v = (r == c) ? -DELTA_F : 0.f;
    KK[(size_t)r * TPd + c] = v;
  }
}

// ---------------- LU: 32x32 diagonal factor + triangular inverses (1 wave) ----------------
__global__ void k_ludiag(float* ws, int base) {
  int b = blockIdx.x, t = threadIdx.x;   // 32 threads
  __shared__ float M[32][33], IL[32][33], IU[32][33];
  float* KK  = wsp(ws, b, OFF_KKT);
  float* gIL = wsp(ws, b, OFF_IL);
  float* gIU = wsp(ws, b, OFF_IU);
  for (int c = 0; c < 32; ++c) M[t][c] = KK[(size_t)(base + t) * TPd + base + c];
  __syncthreads();
  for (int p = 0; p < 32; ++p) {
    if (t == p) {
      float pv = M[p][p];
      if (fabsf(pv) < 1e-20f) M[p][p] = (pv >= 0.f) ? 1e-20f : -1e-20f;
    }
    __syncthreads();
    if (t > p) M[t][p] /= M[p][p];
    __syncthreads();
    if (t > p) {
      float lp = M[t][p];
      for (int c = p + 1; c < 32; ++c) M[t][c] -= lp * M[p][c];
    }
    __syncthreads();
  }
  for (int r = 31; r >= 0; --r) {          // invU, thread t owns column t
    float sum = 0.f;
    if (r <= t) {
      sum = (r == t) ? 1.f : 0.f;
      for (int j = r + 1; j <= t; ++j) sum -= M[r][j] * IU[j][t];
      sum /= M[r][r];
    }
    IU[r][t] = sum;
  }
  for (int r = 0; r < 32; ++r) {           // invL (unit lower), column t
    float sum = 0.f;
    if (r == t) sum = 1.f;
    else if (r > t) { for (int j = t; j < r; ++j) sum -= M[r][j] * IL[j][t]; }
    IL[r][t] = sum;
  }
  __syncthreads();
  for (int c = 0; c < 32; ++c) {
    KK[(size_t)(base + t) * TPd + base + c] = M[t][c];
    gIL[t * 32 + c] = IL[t][c];
    gIU[t * 32 + c] = IU[t][c];
  }
}

// ---------------- LU panels via WMMA: L = A*invU, U = invL*A ----------------
__global__ void k_lupanel(float* ws, int base) {
  int b = blockIdx.y;
  int wave = threadIdx.x >> 5, lane = threadIdx.x & 31;
  int mt = (TPd - base - 32) / 16;
  int tile = blockIdx.x * 8 + wave;
  if (tile >= 4 * mt) return;                   // wave-uniform
  float* KK  = wsp(ws, b, OFF_KKT);
  float* gIL = wsp(ws, b, OFF_IL);
  float* gIU = wsp(ws, b, OFF_IU);
  v8f acc;
  #pragma unroll
  for (int r = 0; r < 8; ++r) acc[r] = 0.f;
  int half = lane >> 4;
  if (tile < 2 * mt) {
    int i0 = base + 32 + (tile >> 1) * 16;
    int c0 = (tile & 1) * 16;
    Frag a  = load_a_g(KK, TPd, i0, base, 1.f, lane);
    Frag bb = load_b_g(gIU, 32, 0, c0, lane);
    acc = wmma3(a, bb, acc);
    int col = base + c0 + (lane & 15);
    #pragma unroll
    for (int r = 0; r < 8; ++r) KK[(size_t)(i0 + r + 8 * half) * TPd + col] = acc[r];
  } else {
    int t2 = tile - 2 * mt;
    int r0 = (t2 & 1) * 16;
    int j0 = base + 32 + (t2 >> 1) * 16;
    Frag a  = load_a_g(gIL, 32, r0, 0, 1.f, lane);
    Frag bb = load_b_g(KK, TPd, base, j0, lane);
    acc = wmma3(a, bb, acc);
    int col = j0 + (lane & 15);
    #pragma unroll
    for (int r = 0; r < 8; ++r) KK[(size_t)(base + r0 + r + 8 * half) * TPd + col] = acc[r];
  }
}

// ---------------- LU trailing update via WMMA: C -= L*U ----------------
// One block per trailing tile-row; L-block staged in LDS, A-frag built once per wave.
__global__ void k_lutrail(float* ws, int base) {
  int b = blockIdx.y;
  int tid = threadIdx.x, wave = tid >> 5, lane = tid & 31;
  int mt = (TPd - base - 32) / 16;
  int i0 = base + 32 + blockIdx.x * 16;
  float* KK = wsp(ws, b, OFF_KKT);
  __shared__ float Lblk[16 * 32];          // [m][k]
  for (int t = tid; t < 16 * 32; t += 256) {
    int r = t >> 5, c = t & 31;
    Lblk[t] = KK[(size_t)(i0 + r) * TPd + base + c];
  }
  __syncthreads();
  Frag a = load_a_lds32(Lblk, -1.f, lane);  // -L (bf16 WMMA has no A-neg modifier)
  int half = lane >> 4;
  for (int jt = wave; jt < mt; jt += 8) {
    int j0 = base + 32 + jt * 16;
    int col = j0 + (lane & 15);
    v8f acc;
    #pragma unroll
    for (int r = 0; r < 8; ++r) acc[r] = KK[(size_t)(i0 + r + 8 * half) * TPd + col];
    Frag bb = load_b_g(KK, TPd, base, j0, lane);
    acc = wmma3(a, bb, acc);
    #pragma unroll
    for (int r = 0; r < 8; ++r) KK[(size_t)(i0 + r + 8 * half) * TPd + col] = acc[r];
  }
}

// ---------------- triangular solves (in-place on rhs) ----------------
__global__ void k_solve(float* ws) {
  int b = blockIdx.x, tid = threadIdx.x;
  float* KK = wsp(ws, b, OFF_KKT);
  float* R  = wsp(ws, b, OFF_RHS);
  for (int p = 0; p < TPd; ++p) {              // forward, unit lower
    float xp = R[p];
    for (int i = p + 1 + tid; i < TPd; i += 256) R[i] -= KK[(size_t)i * TPd + p] * xp;
    __syncthreads();
  }
  for (int p = TPd - 1; p >= 0; --p) {         // backward
    float xp = R[p] / KK[(size_t)p * TPd + p];
    __syncthreads();
    if (tid == 0) R[p] = xp;
    for (int i = tid; i < p; i += 256) R[i] -= KK[(size_t)i * TPd + p] * xp;
    __syncthreads();
  }
}

// ---------------- step: dz/ds, fraction-to-boundary, state update ----------------
__global__ void k_update(const float* __restrict__ G, float* ws) {
  int b = blockIdx.x, tid = threadIdx.x;
  float* X  = wsp(ws, b, OFF_X);
  float* S  = wsp(ws, b, OFF_S);
  float* Z  = wsp(ws, b, OFF_Z);
  float* Y  = wsp(ws, b, OFF_Y);
  float* R3 = wsp(ws, b, OFF_R3);
  float* Tv = wsp(ws, b, OFF_T);
  float* Dv = wsp(ws, b, OFF_D);
  float* RHS = wsp(ws, b, OFF_RHS);            // holds solution
  const float* Gb = G + (size_t)b * NINd * NVARS;
  __shared__ float red[256];
  __shared__ float alps;
  float ds = 0.f, dz = 0.f, ratmin = BIGF;
  int k = tid;
  if (k < NINd) {
    const float* gr = Gb + (size_t)k * NVARS;
    float gdx = 0.f;
    for (int i = 0; i < NVARS; ++i) gdx += gr[i] * RHS[i];
    dz = Dv[k] * gdx + Tv[k];
    ds = -R3[k] - gdx;
    if (ds < 0.f) ratmin = fminf(ratmin, -S[k] / ds);
    if (dz < 0.f) ratmin = fminf(ratmin, -Z[k] / dz);
  }
  red[tid] = ratmin; __syncthreads();
  for (int o = 128; o > 0; o >>= 1) { if (tid < o) red[tid] = fminf(red[tid], red[tid + o]); __syncthreads(); }
  if (tid == 0) alps = fminf(1.f, 0.99f * red[0]);
  __syncthreads();
  float al = alps;
  if (k < NINd) { S[k] += al * ds; Z[k] += al * dz; }
  for (int i = tid; i < NVARS; i += 256) X[i] += al * RHS[i];
  for (int e = tid; e < NEQd; e += 256) Y[e] += al * RHS[NPd + e];
}

// ---------------- extract u0 = x[NX : NX+NU] ----------------
__global__ void k_out(float* ws, float* __restrict__ out) {
  int b = blockIdx.x, t = threadIdx.x;
  if (t < NUd) out[b * NUd + t] = wsp(ws, b, OFF_X)[NXd + t];
}

extern "C" void kernel_launch(void* const* d_in, const int* in_sizes, int n_in,
                              void* d_out, int out_size, void* d_ws, size_t ws_size,
                              hipStream_t stream) {
  (void)in_sizes; (void)n_in; (void)out_size; (void)ws_size;
  const float* w  = (const float*)d_in[0];
  const float* A  = (const float*)d_in[1];
  const float* bv = (const float*)d_in[2];
  const float* G  = (const float*)d_in[3];
  const float* h  = (const float*)d_in[4];
  float* ws  = (float*)d_ws;
  float* out = (float*)d_out;

  k_init<<<NBATCH, 256, 0, stream>>>(w, ws);
  for (int it = 0; it < MAXIT; ++it) {
    k_resid<<<NBATCH, 256, 0, stream>>>(A, bv, G, h, ws);
    k_buildH<<<dim3(HT, NBATCH), 256, 0, stream>>>(G, ws);
    k_fill<<<dim3((NPd * EPd + EPd * TPd + 255) / 256, NBATCH), 256, 0, stream>>>(A, ws);
    for (int kb = 0; kb < NBLK; ++kb) {
      int base = kb * 32;
      k_ludiag<<<NBATCH, 32, 0, stream>>>(ws, base);
      int mt = (TPd - base - 32) / 16;
      if (mt > 0) {
        k_lupanel<<<dim3((4 * mt + 7) / 8, NBATCH), 256, 0, stream>>>(ws, base);
        k_lutrail<<<dim3(mt, NBATCH), 256, 0, stream>>>(ws, base);
      }
    }
    k_solve<<<NBATCH, 256, 0, stream>>>(ws);
    k_update<<<NBATCH, 256, 0, stream>>>(G, ws);
  }
  k_out<<<NBATCH, 32, 0, stream>>>(ws, out);
}
